// CrossAttention_22539988369512
// MI455X (gfx1250) — matrix operand order repaired and checked by reading, not compiled, not run on hardware
//
#include <hip/hip_runtime.h>
#include <hip/hip_bf16.h>

typedef _Float16 half_t;
typedef __attribute__((ext_vector_type(16))) _Float16 v16h;
typedef __attribute__((ext_vector_type(8)))  float    v8f;
typedef __attribute__((ext_vector_type(4)))  unsigned int v4u;
typedef __attribute__((ext_vector_type(4)))  int      v4i;
typedef __attribute__((ext_vector_type(8)))  int      v8i;

#define BATCH   4
#define SEQ_N   4096
#define SEQ_NC  256
#define DIM_C   1024
#define HEADS   16
#define HDIM    64
#define EPS_RMS 1e-6f

// ---------------------------------------------------------------------------
// WMMA fragment helpers (A/B in this code's contiguous k order, C/D per CDNA5 ISA 7.12.2, wave32)
// ---------------------------------------------------------------------------

// A fragment 16x32 f16: lane 0-15: M=lane, K = 0..15 (elts 0-15)
//                       lane 16-31: M=lane-16, K = 16..31
__device__ inline v16h load_a_frag_f32(const float* A, int lda, int m0, int k0, int lane) {
    const int m  = m0 + (lane & 15);
    const int kb = k0 + ((lane >> 4) << 4);
    const float* p = A + (size_t)m * lda + kb;
    v16h a;
#pragma unroll
    for (int i = 0; i < 8; ++i) a[i]     = (half_t)p[i];
#pragma unroll
    for (int i = 0; i < 8; ++i) a[8 + i] = (half_t)p[8 + i];
    return a;
}

__device__ inline v16h load_a_frag_f16(const half_t* A, int lda, int m0, int k0, int lane) {
    const int m  = m0 + (lane & 15);
    const int kb = k0 + ((lane >> 4) << 4);
    const half_t* p = A + (size_t)m * lda + kb;
    v16h a;
#pragma unroll
    for (int i = 0; i < 8; ++i) a[i]     = p[i];
#pragma unroll
    for (int i = 0; i < 8; ++i) a[8 + i] = p[8 + i];
    return a;
}

// B fragment 32x16 from LDS tile of Bt rows: tile is [128 rows][32 k] f16 row-major.
// lane 0-15: N=n_base+lane, K=e ; lane 16-31: N=n_base+lane-16, K=16+e
__device__ inline v16h load_b_frag_lds(const half_t* lb, int n_base, int lane) {
    const int n  = n_base + (lane & 15);
    const int kb = (lane >> 4) << 4;
    const half_t* p = lb + n * 32 + kb;
    v16h b;
#pragma unroll
    for (int e = 0; e < 16; ++e) b[e] = p[e];
    return b;
}

// C/D 16x16 f32: lane 0-15: N=lane, VGPR r -> M=r; lane 16-31: N=lane-16, M=8+r
__device__ inline void store_d_frag(float* C, int ldc, int m0, int n0, int lane,
                                    v8f d, const float* bias) {
    const int n  = n0 + (lane & 15);
    const int mb = m0 + ((lane >> 4) << 3);
    const float badd = bias ? bias[n] : 0.0f;
#pragma unroll
    for (int r = 0; r < 8; ++r)
        C[(size_t)(mb + r) * ldc + n] = d[r] + badd;
}

// ---------------------------------------------------------------------------
// TDM: async 2-D f16 tile (tile_d0 elems/row x tile_d1 rows) global -> LDS.
// Descriptor packing per CDNA5 ISA (D# groups 0/1).
// ---------------------------------------------------------------------------
__device__ inline void tdm_load_2d_f16(unsigned lds_byte_addr, const half_t* gptr,
                                       unsigned tensor_d0, unsigned tensor_d1,
                                       unsigned tile_d0, unsigned tile_d1) {
    const unsigned long long ga = (unsigned long long)(uintptr_t)gptr;
    v4u g0;
    g0[0] = 1u;                                           // count=1, user descriptor
    g0[1] = lds_byte_addr;                                // LDS dest (bytes)
    g0[2] = (unsigned)(ga & 0xFFFFFFFFu);                 // global_addr[31:0]
    g0[3] = (unsigned)((ga >> 32) & 0x01FFFFFFu)          // global_addr[56:32]
          | (2u << 30);                                   // type = 2 ("image")
    v8i g1;
    g1[0] = (int)(1u << 16);                              // wg_mask=0, data_size=1 (2B)
    g1[1] = (int)((tensor_d0 & 0xFFFFu) << 16);           // tensor_dim0[15:0]
    g1[2] = (int)(((tensor_d0 >> 16) & 0xFFFFu)
          |       ((tensor_d1 & 0xFFFFu) << 16));         // dim0[31:16] | dim1[15:0]
    g1[3] = (int)(((tensor_d1 >> 16) & 0xFFFFu)
          |       ((tile_d0 & 0xFFFFu) << 16));           // dim1[31:16] | tile_dim0
    g1[4] = (int)(tile_d1 & 0xFFFFu);                     // tile_dim1 (tile_dim2 = 0)
    g1[5] = (int)tensor_d0;                               // tensor_dim0_stride[31:0]
    g1[6] = 0;                                            // stride0[47:32] | stride1[15:0]
    g1[7] = 0;
    const v4i z4 = {0, 0, 0, 0};
#if defined(__clang_major__) && (__clang_major__ >= 23)
    const v8i z8 = {0, 0, 0, 0, 0, 0, 0, 0};
    __builtin_amdgcn_tensor_load_to_lds(g0, g1, z4, z4, z8, 0);
#else
    __builtin_amdgcn_tensor_load_to_lds(g0, g1, z4, z4, 0);
#endif
}

// ---------------------------------------------------------------------------
// Transpose + convert: dst[n*K + k] = (f16) src[k*N + n]
// ---------------------------------------------------------------------------
__global__ void transpose_cvt_kernel(const float* __restrict__ src,
                                     half_t* __restrict__ dst, int K, int N) {
    int idx = blockIdx.x * blockDim.x + threadIdx.x;
    if (idx >= K * N) return;
    const int k = idx % K;
    const int n = idx / K;
    dst[idx] = (half_t)src[(size_t)k * N + n];
}

// ---------------------------------------------------------------------------
// WMMA GEMM with double-buffered TDM-staged B:
//   C[M,N] = A[M,K] * Bt[N,K]^T (+ bias)
// block = 128 threads (4 waves, 2x2), block tile 128x128, wave tile 64x64
// (4x4 fragments). B k-slabs (128 rows x 32 k, f16 = 8KB) ping-pong through
// LDS via the Tensor Data Mover; slab i+1 streams while slab i feeds WMMAs
// (TDM ops from one wave complete in-order -> s_wait_tensorcnt<=1 suffices).
// ---------------------------------------------------------------------------
template <bool A_IS_F32>
__global__ __launch_bounds__(128)
void wmma_gemm_kernel(const void* __restrict__ Ap, const half_t* __restrict__ Bt,
                      float* __restrict__ C, int M, int N, int K,
                      const float* __restrict__ bias) {
    __shared__ half_t lds_b[2][128 * 32];

    const int lane = threadIdx.x & 31;
    const int wave = threadIdx.x >> 5;
    const int m0 = blockIdx.y * 128 + (wave >> 1) * 64;
    const int nb = (wave & 1) * 64;                 // wave's N offset inside block tile
    const int n0 = blockIdx.x * 128 + nb;

    const unsigned lds_base[2] = {
        (unsigned)(unsigned long long)(uintptr_t)&lds_b[0][0],
        (unsigned)(unsigned long long)(uintptr_t)&lds_b[1][0]
    };
    const half_t* brow = Bt + (size_t)blockIdx.x * 128 * K;

    v8f acc[4][4] = {};

    // prologue: start DMA of first B slab
    if (threadIdx.x == 0)
        tdm_load_2d_f16(lds_base[0], brow, (unsigned)K, (unsigned)N, 32u, 128u);

    int cur = 0;
    for (int k0 = 0; k0 < K; k0 += 32, cur ^= 1) {
        if (threadIdx.x == 0) {
            if (k0 + 32 < K) {
                // stream next slab into the other buffer, then wait for current
                tdm_load_2d_f16(lds_base[cur ^ 1], brow + (k0 + 32),
                                (unsigned)K, (unsigned)N, 32u, 128u);
                __builtin_amdgcn_s_wait_tensorcnt(1);
            } else {
                __builtin_amdgcn_s_wait_tensorcnt(0);
            }
        }
        __syncthreads();   // publish current slab to all waves

        if (k0 + 32 < K)   // prefetch next A slab (gfx1250 global_prefetch)
            __builtin_prefetch((const char*)Ap + ((size_t)(m0 + (lane & 15)) * K + k0 + 32)
                                               * (A_IS_F32 ? 4 : 2), 0, 1);

        v16h a[4];
#pragma unroll
        for (int i = 0; i < 4; ++i) {
            if constexpr (A_IS_F32)
                a[i] = load_a_frag_f32((const float*)Ap, K, m0 + 16 * i, k0, lane);
            else
                a[i] = load_a_frag_f16((const half_t*)Ap, K, m0 + 16 * i, k0, lane);
        }
#pragma unroll
        for (int j = 0; j < 4; ++j) {
            const v16h bf = load_b_frag_lds(&lds_b[cur][0], nb + 16 * j, lane);
#pragma unroll
            for (int i = 0; i < 4; ++i)
                acc[i][j] = __builtin_amdgcn_wmma_f32_16x16x32_f16(
                    false, a[i], false, bf, (short)0, acc[i][j], false, false);
        }
        __syncthreads();   // all reads of lds_b[cur] done before it is re-filled
    }

#pragma unroll
    for (int i = 0; i < 4; ++i)
#pragma unroll
        for (int j = 0; j < 4; ++j)
            store_d_frag(C, N, m0 + 16 * i, n0 + 16 * j, lane, acc[i][j], bias);
}

// ---------------------------------------------------------------------------
// Q: RMS-norm over head dim + RoPE, fp32 [B,N,H*D] -> f16 [B*H, N, D]
// ---------------------------------------------------------------------------
__global__ void q_norm_rope_kernel(const float* __restrict__ q,
                                   const float* __restrict__ cosb,
                                   const float* __restrict__ sinb,
                                   const float* __restrict__ w,
                                   half_t* __restrict__ qh) {
    int idx = blockIdx.x * blockDim.x + threadIdx.x;   // (b, n, h)
    if (idx >= BATCH * SEQ_N * HEADS) return;
    const int h = idx % HEADS;
    const int n = (idx / HEADS) % SEQ_N;
    const int b = idx / (HEADS * SEQ_N);

    const float* src = q + ((size_t)b * SEQ_N + n) * DIM_C + h * HDIM;
    float v[HDIM];
    float ss = 0.0f;
#pragma unroll
    for (int d = 0; d < HDIM; ++d) { v[d] = src[d]; ss += v[d] * v[d]; }
    const float inv = rsqrtf(ss * (1.0f / HDIM) + EPS_RMS);

    const float* cs = cosb + (size_t)n * HDIM;
    const float* sn = sinb + (size_t)n * HDIM;
    half_t* dst = qh + (((size_t)b * HEADS + h) * SEQ_N + n) * HDIM;
#pragma unroll
    for (int d = 0; d < HDIM; ++d) {
        const float nv  = v[d] * inv * w[d];
        const float rot = (d < HDIM / 2) ? -(v[d + HDIM / 2] * inv * w[d + HDIM / 2])
                                         :  (v[d - HDIM / 2] * inv * w[d - HDIM / 2]);
        dst[d] = (half_t)(nv * cs[d] + rot * sn[d]);
    }
}

// K: RMS-norm, fp32 [B,Nc,H*D] -> f16 [B*H, Nc, D]
__global__ void k_norm_kernel(const float* __restrict__ k,
                              const float* __restrict__ w,
                              half_t* __restrict__ kh) {
    int idx = blockIdx.x * blockDim.x + threadIdx.x;   // (b, nc, h)
    if (idx >= BATCH * SEQ_NC * HEADS) return;
    const int h  = idx % HEADS;
    const int nc = (idx / HEADS) % SEQ_NC;
    const int b  = idx / (HEADS * SEQ_NC);

    const float* src = k + ((size_t)b * SEQ_NC + nc) * DIM_C + h * HDIM;
    float v[HDIM];
    float ss = 0.0f;
#pragma unroll
    for (int d = 0; d < HDIM; ++d) { v[d] = src[d]; ss += v[d] * v[d]; }
    const float inv = rsqrtf(ss * (1.0f / HDIM) + EPS_RMS);

    half_t* dst = kh + (((size_t)b * HEADS + h) * SEQ_NC + nc) * HDIM;
#pragma unroll
    for (int d = 0; d < HDIM; ++d) dst[d] = (half_t)(v[d] * inv * w[d]);
}

// V: reorder + convert, fp32 [B,Nc,H*D] -> f16 [B*H, Nc, D]
__global__ void v_reorder_kernel(const float* __restrict__ v, half_t* __restrict__ vh) {
    int idx = blockIdx.x * blockDim.x + threadIdx.x;
    if (idx >= BATCH * SEQ_NC * DIM_C) return;
    const int col = idx % DIM_C;
    const int nc  = (idx / DIM_C) % SEQ_NC;
    const int b   = idx / (DIM_C * SEQ_NC);
    const int h = col / HDIM, d = col % HDIM;
    vh[(((size_t)b * HEADS + h) * SEQ_NC + nc) * HDIM + d] = (half_t)v[idx];
}

// ---------------------------------------------------------------------------
// Attention: one wave per 16 query rows of one (b,h).
//   S = Q K^T (WMMA) -> LDS, softmax (half-wave shfl reductions),
//   P (f16, LDS) -> O = P V (WMMA), scale by 1/rowsum on store.
// ---------------------------------------------------------------------------
#define ATT_WAVES 2

__global__ __launch_bounds__(32 * ATT_WAVES)
void attn_kernel(const half_t* __restrict__ qh, const half_t* __restrict__ kh,
                 const half_t* __restrict__ vh, half_t* __restrict__ oh) {
    __shared__ float  s_s[ATT_WAVES][16][SEQ_NC];
    __shared__ half_t s_p[ATT_WAVES][16][SEQ_NC];
    __shared__ float  s_inv[ATT_WAVES][16];

    const int lane = threadIdx.x & 31;
    const int w    = threadIdx.x >> 5;
    const int tile = blockIdx.x * ATT_WAVES + w;   // 0 .. B*H*(N/16)-1
    const int nt   = tile & (SEQ_N / 16 - 1);
    const int bh   = tile >> 8;                    // N/16 == 256
    const int b    = bh >> 4;                      // H == 16
    const int h    = bh & (HEADS - 1);

    const half_t* qb = qh + ((size_t)bh * SEQ_N + nt * 16) * HDIM;
    const half_t* kb = kh + (size_t)bh * SEQ_NC * HDIM;
    const half_t* vb = vh + (size_t)bh * SEQ_NC * HDIM;

    const int   mrow  = lane & 15;
    const int   hsel  = lane >> 4;
    const float scale = 0.125f;   // HDIM^-0.5

    // ---- Phase 1: S = scale * Q K^T ----
    for (int kt = 0; kt < SEQ_NC / 16; ++kt) {
        v8f acc = {};
#pragma unroll
        for (int ks = 0; ks < HDIM; ks += 32) {
            v16h a;
            {
                const half_t* p = qb + (size_t)mrow * HDIM + ks + hsel * 16;
#pragma unroll
                for (int i = 0; i < 8; ++i) a[i]     = p[i];
#pragma unroll
                for (int i = 0; i < 8; ++i) a[8 + i] = p[8 + i];
            }
            v16h bf;   // B[k=d][n=key] = K[key][d] — contiguous in d per lane
            {
                const half_t* p = kb + (size_t)(kt * 16 + mrow) * HDIM + ks + hsel * 16;
#pragma unroll
                for (int e = 0; e < 16; ++e) bf[e] = p[e];
            }
            acc = __builtin_amdgcn_wmma_f32_16x16x32_f16(false, a, false, bf, (short)0, acc, false, false);
        }
#pragma unroll
        for (int r = 0; r < 8; ++r)
            s_s[w][hsel * 8 + r][kt * 16 + mrow] = acc[r] * scale;
    }
    __syncthreads();

    // ---- Phase 2: softmax (unnormalized exp + row sums) ----
    {
        const int row = mrow;
        const int cb  = hsel * (SEQ_NC / 2);
        float mx = -3.0e38f;
        for (int j = 0; j < SEQ_NC / 2; ++j) mx = fmaxf(mx, s_s[w][row][cb + j]);
        mx = fmaxf(mx, __shfl_xor(mx, 16, 32));
        float sum = 0.0f;
        for (int j = 0; j < SEQ_NC / 2; ++j) {
            const float e = __expf(s_s[w][row][cb + j] - mx);
            sum += e;
            s_p[w][row][cb + j] = (half_t)e;
        }
        sum += __shfl_xor(sum, 16, 32);
        if (hsel == 0) s_inv[w][row] = 1.0f / sum;
    }
    __syncthreads();

    // ---- Phase 3: O = P V ----
    v8f oacc[4] = {};
    for (int kt = 0; kt < SEQ_NC / 32; ++kt) {
        v16h a;
        {
            const int kbase = kt * 32 + hsel * 16;
#pragma unroll
            for (int i = 0; i < 8; ++i) a[i]     = s_p[w][mrow][kbase + i];
#pragma unroll
            for (int i = 0; i < 8; ++i) a[8 + i] = s_p[w][mrow][kbase + 8 + i];
        }
#pragma unroll
        for (int j = 0; j < 4; ++j) {
            v16h bf;
            const int kbB = kt * 32 + hsel * 16;
            const half_t* p = vb + (size_t)kbB * HDIM + j * 16 + mrow;
#pragma unroll
            for (int e = 0; e < 16; ++e) bf[e] = p[(size_t)e * HDIM];
            oacc[j] = __builtin_amdgcn_wmma_f32_16x16x32_f16(false, a, false, bf, (short)0, oacc[j], false, false);
        }
    }

    // ---- Phase 4: store O (f16) into [B, N, DIM] with col = h*64+d ----
#pragma unroll
    for (int j = 0; j < 4; ++j) {
#pragma unroll
        for (int r = 0; r < 8; ++r) {
            const int lrow = hsel * 8 + r;
            const size_t row = (size_t)b * SEQ_N + nt * 16 + lrow;
            oh[row * DIM_C + h * HDIM + j * 16 + mrow] =
                (half_t)(oacc[j][r] * s_inv[w][lrow]);
        }
    }
}

// ---------------------------------------------------------------------------
// Host orchestration
// ---------------------------------------------------------------------------
extern "C" void kernel_launch(void* const* d_in, const int* in_sizes, int n_in,
                              void* d_out, int out_size, void* d_ws, size_t ws_size,
                              hipStream_t stream) {
    const float* x  = (const float*)d_in[0];
    const float* c  = (const float*)d_in[1];
    const float* rc = (const float*)d_in[2];
    const float* rs = (const float*)d_in[3];
    const float* Wq = (const float*)d_in[4];
    const float* Wk = (const float*)d_in[5];
    const float* Wv = (const float*)d_in[6];
    const float* Wo = (const float*)d_in[7];
    const float* bo = (const float*)d_in[8];
    const float* qw = (const float*)d_in[9];
    const float* kw = (const float*)d_in[10];
    float* out = (float*)d_out;

    char* ws = (char*)d_ws;
    auto alloc = [&](size_t bytes) -> void* {
        void* p = (void*)ws;
        ws += (bytes + 255) & ~(size_t)255;
        return p;
    };
    const size_t WE = (size_t)DIM_C * DIM_C;
    half_t* WqT = (half_t*)alloc(WE * 2);   // transposed f16 weights [N][K]
    half_t* WkT = (half_t*)alloc(WE * 2);
    half_t* WvT = (half_t*)alloc(WE * 2);
    half_t* WoT = (half_t*)alloc(WE * 2);
    float*  q32 = (float*)alloc((size_t)BATCH * SEQ_N  * DIM_C * 4);
    float*  k32 = (float*)alloc((size_t)BATCH * SEQ_NC * DIM_C * 4);
    float*  v32 = (float*)alloc((size_t)BATCH * SEQ_NC * DIM_C * 4);
    half_t* q_h = (half_t*)alloc((size_t)BATCH * SEQ_N  * DIM_C * 2);
    half_t* k_h = (half_t*)alloc((size_t)BATCH * SEQ_NC * DIM_C * 2);
    half_t* v_h = (half_t*)alloc((size_t)BATCH * SEQ_NC * DIM_C * 2);
    half_t* o_h = (half_t*)alloc((size_t)BATCH * SEQ_N  * DIM_C * 2);

    // 1) weights -> transposed f16
    {
        const int n = (int)WE, blk = 256, grd = (n + blk - 1) / blk;
        transpose_cvt_kernel<<<grd, blk, 0, stream>>>(Wq, WqT, DIM_C, DIM_C);
        transpose_cvt_kernel<<<grd, blk, 0, stream>>>(Wk, WkT, DIM_C, DIM_C);
        transpose_cvt_kernel<<<grd, blk, 0, stream>>>(Wv, WvT, DIM_C, DIM_C);
        transpose_cvt_kernel<<<grd, blk, 0, stream>>>(Wo, WoT, DIM_C, DIM_C);
    }

    // 2) projections (WMMA GEMMs, double-buffered TDM-staged B)
    const dim3 blk128(128);
    wmma_gemm_kernel<true><<<dim3(DIM_C / 128, BATCH * SEQ_N / 128), blk128, 0, stream>>>(
        x, WqT, q32, BATCH * SEQ_N, DIM_C, DIM_C, nullptr);
    wmma_gemm_kernel<true><<<dim3(DIM_C / 128, BATCH * SEQ_NC / 128), blk128, 0, stream>>>(
        c, WkT, k32, BATCH * SEQ_NC, DIM_C, DIM_C, nullptr);
    wmma_gemm_kernel<true><<<dim3(DIM_C / 128, BATCH * SEQ_NC / 128), blk128, 0, stream>>>(
        c, WvT, v32, BATCH * SEQ_NC, DIM_C, DIM_C, nullptr);

    // 3) norms / rope / reorder -> f16 heads
    q_norm_rope_kernel<<<(BATCH * SEQ_N * HEADS + 255) / 256, 256, 0, stream>>>(q32, rc, rs, qw, q_h);
    k_norm_kernel<<<(BATCH * SEQ_NC * HEADS + 255) / 256, 256, 0, stream>>>(k32, kw, k_h);
    v_reorder_kernel<<<(BATCH * SEQ_NC * DIM_C + 255) / 256, 256, 0, stream>>>(v32, v_h);

    // 4) attention
    attn_kernel<<<(BATCH * HEADS * (SEQ_N / 16)) / ATT_WAVES, 32 * ATT_WAVES, 0, stream>>>(
        q_h, k_h, v_h, o_h);

    // 5) output projection + bias -> fp32 out
    wmma_gemm_kernel<false><<<dim3(DIM_C / 128, BATCH * SEQ_N / 128), blk128, 0, stream>>>(
        o_h, WoT, out, BATCH * SEQ_N, DIM_C, DIM_C, bo);
}
